// Sparsifier_26706106646977
// MI455X (gfx1250) — matrix operand order repaired and checked by reading, not compiled, not run on hardware
//
#include <hip/hip_runtime.h>
#include <hip/hip_bf16.h>
#include <math.h>

typedef __bf16 bf16;
typedef __attribute__((ext_vector_type(16))) __bf16 v16bf;
typedef __attribute__((ext_vector_type(8)))  __bf16 v8bf;
typedef __attribute__((ext_vector_type(8)))  float  v8f;
typedef int v4i __attribute__((vector_size(4 * sizeof(int))));

#define H 256
#define W 256
#define B_N 8
#define D_N 64
#define KTAPS 1089     // 33*33
#define KPAD 1120      // 35*32

// gfx1250 async Global->LDS path (ASYNCcnt), guarded for toolchain portability
#if defined(__has_builtin)
#if __has_builtin(__builtin_amdgcn_global_load_async_to_lds_b128) && \
    __has_builtin(__builtin_amdgcn_s_wait_asynccnt)
#define USE_ASYNC_LDS 1
#endif
#endif

// ---------------------------------------------------------------------------
// Prep: split fp32 input into bf16 hi/lo, init use_map border mask + scale
// ---------------------------------------------------------------------------
__global__ void split_input_kernel(const float* __restrict__ in,
                                   bf16* __restrict__ hi, bf16* __restrict__ lo,
                                   float* __restrict__ use_map,
                                   float* __restrict__ scale) {
    int i = blockIdx.x * 256 + threadIdx.x;      // 8*65536 total, grid exact
    float v = in[i];
    bf16 h = (bf16)v;
    hi[i] = h;
    lo[i] = (bf16)(v - (float)h);
    int hw = i & 65535;
    int hr = hw >> 8, wc = hw & 255;
    use_map[i] = (hr < 16 || hr >= 240 || wc < 16 || wc >= 240) ? 0.0f : 1.0f;
    if (i < B_N) scale[i] = 1.0f;
}

// ---------------------------------------------------------------------------
// Prep: split filter into bf16 hi/lo, pad K 1089 -> 1120 with zeros
// ---------------------------------------------------------------------------
__global__ void split_filter_kernel(const float* __restrict__ filt,
                                    bf16* __restrict__ hi, bf16* __restrict__ lo) {
    int i = blockIdx.x * 256 + threadIdx.x;      // 64*1120 total, grid exact
    int atom = i / KPAD;
    int k = i - atom * KPAD;
    float v = (k < KTAPS) ? filt[atom * KTAPS + k] : 0.0f;
    bf16 h = (bf16)v;
    hi[i] = h;
    lo[i] = (bf16)(v - (float)h);
}

// ---------------------------------------------------------------------------
// Implicit-GEMM circular conv via 3xBF16-split WMMA.
// Block = 128 threads = 4 waves. Block covers (b, x, y0..y0+15) x all 64 atoms.
// Wave w computes D-tile: A = filt[16w..16w+15][K]  (16x32 bf16 per step)
//                         B = patch[K][16 pixels]   (32x16 bf16 per step)
// Patch staged as 33 full-width rows in LDS via async Global->LDS b128 copies.
// ---------------------------------------------------------------------------
__global__ __launch_bounds__(128) void conv_wmma_kernel(
    const bf16* __restrict__ inHi, const bf16* __restrict__ inLo,
    const bf16* __restrict__ fHi,  const bf16* __restrict__ fLo,
    float* __restrict__ conv)
{
    __shared__ __align__(16) bf16 sHi[33 * 256];
    __shared__ __align__(16) bf16 sLo[33 * 256];

    int blk = blockIdx.x;              // 8*256*16 = 32768 blocks
    int b   = blk >> 12;
    int rem = blk & 4095;
    int x   = rem >> 4;
    int y0  = (rem & 15) << 4;

    int tid = threadIdx.x;
    const bf16* inHb = inHi + b * (H * W);
    const bf16* inLb = inLo + b * (H * W);

    // Stage 33 full (wrapped) rows; each chunk = 8 bf16 = 16 bytes.
    // 33 rows * 32 chunks = 1056 chunks per array.
    for (int c = tid; c < 33 * 32; c += 128) {
        int row  = c >> 5;
        int cu   = (c & 31) << 3;              // element offset in row
        int grow = (x + row - 16) & 255;
        int ge   = grow * 256 + cu;            // global element offset
        int le   = row * 256 + cu;             // LDS element offset
#ifdef USE_ASYNC_LDS
        __builtin_amdgcn_global_load_async_to_lds_b128(
            (__attribute__((address_space(1))) v4i*)(inHb + ge),
            (__attribute__((address_space(3))) v4i*)(&sHi[le]), 0, 0);
        __builtin_amdgcn_global_load_async_to_lds_b128(
            (__attribute__((address_space(1))) v4i*)(inLb + ge),
            (__attribute__((address_space(3))) v4i*)(&sLo[le]), 0, 0);
#else
        *(uint4*)&sHi[le] = *(const uint4*)(inHb + ge);
        *(uint4*)&sLo[le] = *(const uint4*)(inLb + ge);
#endif
    }
#ifdef USE_ASYNC_LDS
    __builtin_amdgcn_s_wait_asynccnt(0);
#endif
    __syncthreads();

    int lane   = tid & 31;
    int wv     = tid >> 5;             // wave id 0..3 -> atom group
    int upper  = lane >> 4;            // lane >= 16 ?
    int n      = lane & 15;            // A: atom row   B: pixel col

    // A (filter) per-lane base: lane<16 -> K {0..7,16..23}; lane>=16 -> +8
    int atom = wv * 16 + n;
    const bf16* fHb = fHi + atom * KPAD + (upper ? 8 : 0);
    const bf16* fLb = fLo + atom * KPAD + (upper ? 8 : 0);
    int kbB  = upper ? 8 : 0;          // B: lanes 16-31 hold K+8 (same as A)
    int col0 = y0 + n - 16;            // base column for this pixel (pre-wrap)

    v8f acc = {0.f, 0.f, 0.f, 0.f, 0.f, 0.f, 0.f, 0.f};

    #pragma unroll 5
    for (int k0 = 0; k0 < KPAD; k0 += 32) {
        // Prefetch next filter chunk (streams through L2)
        __builtin_prefetch(fHb + k0 + 64, 0, 1);

        // A fragments: two contiguous 16B loads each (exact WMMA A layout)
        v8bf ah0 = *(const v8bf*)(fHb + k0);
        v8bf ah1 = *(const v8bf*)(fHb + k0 + 16);
        v8bf al0 = *(const v8bf*)(fLb + k0);
        v8bf al1 = *(const v8bf*)(fLb + k0 + 16);
        v16bf aH = __builtin_shufflevector(ah0, ah1, 0,1,2,3,4,5,6,7,8,9,10,11,12,13,14,15);
        v16bf aL = __builtin_shufflevector(al0, al1, 0,1,2,3,4,5,6,7,8,9,10,11,12,13,14,15);

        // B fragments from LDS: lane holds K {0..7,16..23} (+8 upper) taps of pixel n
        v16bf bHf, bLf;
        #pragma unroll
        for (int e = 0; e < 16; ++e) {
            int k = k0 + kbB + (e & 7) + ((e >> 3) << 4);
            bf16 vh = (bf16)0.0f, vl = (bf16)0.0f;
            if (k < KTAPS) {
                int p = k / 33;
                int q = k - p * 33;
                int a = p * 256 + ((col0 + q) & 255);
                vh = sHi[a];
                vl = sLo[a];
            }
            bHf[e] = vh;
            bLf[e] = vl;
        }
        // fp32 ~= hi*hi + hi*lo + lo*hi  (drop lo*lo)
        acc = __builtin_amdgcn_wmma_f32_16x16x32_bf16(false, aH, false, bHf, (short)0, acc, false, false);
        acc = __builtin_amdgcn_wmma_f32_16x16x32_bf16(false, aH, false, bLf, (short)0, acc, false, false);
        acc = __builtin_amdgcn_wmma_f32_16x16x32_bf16(false, aL, false, bHf, (short)0, acc, false, false);
    }

    // D layout: VGPR r, lane<16 -> M=r ; lane>=16 -> M=r+8 ; N = lane&15
    int atomBase = wv * 16 + (upper ? 8 : 0);
    float* outBase = conv + (size_t)b * D_N * H * W + (size_t)x * W + y0 + n;
    #pragma unroll
    for (int r = 0; r < 8; ++r) {
        outBase[(size_t)(atomBase + r) * (H * W)] = acc[r];
    }
}

// ---------------------------------------------------------------------------
// Per-batch argmax of conv * use_raw * scale * prior (one block per batch).
// conv (134 MB) is L2-resident on MI455X (192 MB L2).
// ---------------------------------------------------------------------------
__global__ __launch_bounds__(1024) void argmax_kernel(
    const float* __restrict__ conv, const float* __restrict__ use_map,
    const float* __restrict__ prior, const float* __restrict__ scale,
    float* __restrict__ outp, int* __restrict__ centers, int t)
{
    __shared__ float sv[1024];
    __shared__ int   si[1024];
    int b = blockIdx.x;
    int tid = threadIdx.x;
    const float* cb = conv + (size_t)b * D_N * H * W;
    const float* ub = use_map + b * (H * W);
    float sc = scale[b];

    float best = -INFINITY;
    int   bidx = 0;
    for (int i = tid; i < D_N * H * W; i += 1024) {
        float v = cb[i] * ub[i & 65535] * prior[i >> 16] * sc;
        if (v > best || (v == best && i < bidx)) { best = v; bidx = i; }
    }
    sv[tid] = best; si[tid] = bidx;
    __syncthreads();
    for (int s = 512; s > 0; s >>= 1) {
        if (tid < s) {
            float ov = sv[tid + s]; int oi = si[tid + s];
            if (ov > sv[tid] || (ov == sv[tid] && oi < si[tid])) { sv[tid] = ov; si[tid] = oi; }
        }
        __syncthreads();
    }
    if (tid == 0) {
        int idx = si[0];
        int d  = idx >> 16;
        int hw = idx & 65535;
        int px = hw >> 8, py = hw & 255;
        // position_found [8,10,3] then overlap_found [8,10,2], flat
        outp[b * 30 + t * 3 + 0] = (float)d;
        outp[b * 30 + t * 3 + 1] = (float)px;
        outp[b * 30 + t * 3 + 2] = (float)py;
        outp[240 + b * 20 + t * 2 + 0] = sv[0];     // ov0: max incl. prior & norm
        outp[240 + b * 20 + t * 2 + 1] = cb[idx];   // ov1: raw overlap
        centers[b * 2]     = px;
        centers[b * 2 + 1] = py;
    }
}

// ---------------------------------------------------------------------------
// Deadzone update: use *= (1 - exp(-d2/32)) * (d2>16), track new max,
// carry normalization as scale = 1/max (argmax is scale-invariant).
// ---------------------------------------------------------------------------
__global__ __launch_bounds__(1024) void update_kernel(
    float* __restrict__ use_map, const int* __restrict__ centers,
    float* __restrict__ scale)
{
    __shared__ float sm[1024];
    int b = blockIdx.x;
    int tid = threadIdx.x;
    int px = centers[b * 2], py = centers[b * 2 + 1];
    float* ub = use_map + b * (H * W);
    float lmax = 0.0f;
    for (int hw = tid; hw < H * W; hw += 1024) {
        int hr = hw >> 8, wc = hw & 255;
        float dx = (float)(hr - px);
        float dy = (float)(wc - py);
        float d2 = dx * dx + dy * dy;
        float tmap = 1.0f - expf(-d2 * (1.0f / 32.0f));   // 2*sigma^2 = 32
        if (d2 <= 16.0f) tmap = 0.0f;                     // hard cutout r=4
        float nu = ub[hw] * tmap;
        ub[hw] = nu;
        lmax = fmaxf(lmax, nu);
    }
    sm[tid] = lmax;
    __syncthreads();
    for (int s = 512; s > 0; s >>= 1) {
        if (tid < s) sm[tid] = fmaxf(sm[tid], sm[tid + s]);
        __syncthreads();
    }
    if (tid == 0) scale[b] = 1.0f / sm[0];
}

// ---------------------------------------------------------------------------
extern "C" void kernel_launch(void* const* d_in, const int* in_sizes, int n_in,
                              void* d_out, int out_size, void* d_ws, size_t ws_size,
                              hipStream_t stream) {
    const float* input  = (const float*)d_in[0];
    const float* dfilt  = (const float*)d_in[1];
    // d_in[2] (dictionary) unused by reference
    const float* prior  = (const float*)d_in[3];
    float* outp = (float*)d_out;

    // Workspace carve (256B aligned)
    char* ws = (char*)d_ws;
    size_t off = 0;
    auto carve = [&](size_t bytes) -> char* {
        char* p = ws + off;
        off += (bytes + 255) & ~(size_t)255;
        return p;
    };
    float* conv    = (float*)carve((size_t)B_N * D_N * H * W * 4);  // 134 MB
    bf16*  inHi    = (bf16*) carve((size_t)B_N * H * W * 2);
    bf16*  inLo    = (bf16*) carve((size_t)B_N * H * W * 2);
    bf16*  fHi     = (bf16*) carve((size_t)D_N * KPAD * 2);
    bf16*  fLo     = (bf16*) carve((size_t)D_N * KPAD * 2);
    float* use_map = (float*)carve((size_t)B_N * H * W * 4);
    float* scale   = (float*)carve(B_N * 4);
    int*   centers = (int*)  carve(B_N * 2 * 4);
    (void)ws_size; (void)in_sizes; (void)n_in; (void)out_size;

    split_input_kernel<<<(B_N * H * W) / 256, 256, 0, stream>>>(
        input, inHi, inLo, use_map, scale);
    split_filter_kernel<<<(D_N * KPAD) / 256, 256, 0, stream>>>(
        dfilt, fHi, fLo);
    conv_wmma_kernel<<<B_N * H * (W / 16), 128, 0, stream>>>(
        inHi, inLo, fHi, fLo, conv);

    for (int t = 0; t < 10; ++t) {
        argmax_kernel<<<B_N, 1024, 0, stream>>>(
            conv, use_map, prior, scale, outp, centers, t);
        update_kernel<<<B_N, 1024, 0, stream>>>(
            use_map, centers, scale);
    }
}
